// LSTMModel_33638183862804
// MI455X (gfx1250) — matrix operand order repaired and checked by reading, not compiled, not run on hardware
//
#include <hip/hip_runtime.h>

#define N_BATCH 128
#define T_STEPS 256
#define IN_DIM  768          // 16*32 + 8*32
#define H_DIM   1024
#define GATES   4096         // 4*H
#define KT_L0   56           // (1024+768)/32
#define KT_L1   64           // 2048/32
#define NT_TILES 256         // 4096/16

typedef __attribute__((ext_vector_type(16))) __bf16       v16bf;
typedef __attribute__((ext_vector_type(8)))  float        v8f;
typedef __attribute__((ext_vector_type(8)))  unsigned int v8u;
typedef __attribute__((ext_vector_type(4)))  unsigned int v4u;

union BfPack { v8u u; v16bf b; };

__device__ __forceinline__ unsigned short f2bf(float f) {
  unsigned int u = __float_as_uint(f);
  u += 0x7FFFu + ((u >> 16) & 1u);               // round to nearest even
  return (unsigned short)(u >> 16);
}

// ---------------------------------------------------------------------------
// Feature-wise batchnorm stats over N*T rows: stats[f]=mean, stats[16+f]=rstd
// ---------------------------------------------------------------------------
__global__ __launch_bounds__(256)
void bn_stats(const float* __restrict__ x, float* __restrict__ stats) {
  const int f = blockIdx.x;              // 0..15
  const int tid = threadIdx.x;
  __shared__ float red[256];
  float s = 0.f, s2 = 0.f;
  for (int i = tid; i < N_BATCH * T_STEPS; i += 256) {
    float v = x[(size_t)i * 24 + f];
    s += v; s2 += v * v;
  }
  red[tid] = s; __syncthreads();
  for (int o = 128; o > 0; o >>= 1) { if (tid < o) red[tid] += red[tid + o]; __syncthreads(); }
  float mean = red[0] * (1.f / (N_BATCH * T_STEPS));
  __syncthreads();
  red[tid] = s2; __syncthreads();
  for (int o = 128; o > 0; o >>= 1) { if (tid < o) red[tid] += red[tid + o]; __syncthreads(); }
  if (tid == 0) {
    float var = red[0] * (1.f / (N_BATCH * T_STEPS)) - mean * mean;
    stats[f]      = mean;
    stats[16 + f] = rsqrtf(var + 1e-5f);
  }
}

// ---------------------------------------------------------------------------
// feats (N,T,768) bf16: numeric BN*outer(QnV) + categorical QlV gather
// ---------------------------------------------------------------------------
__global__ __launch_bounds__(256)
void build_feats(const float* __restrict__ x, const float* __restrict__ QnV,
                 const float* __restrict__ QlV, const float* __restrict__ bng,
                 const float* __restrict__ bnb, const float* __restrict__ stats,
                 unsigned short* __restrict__ feats) {
  const size_t i = blockIdx.x;           // n*T + t, 0..32767
  const int tid = threadIdx.x;
  #pragma unroll
  for (int jj = 0; jj < 3; ++jj) {
    const int j = tid + jj * 256;        // 0..767
    float val;
    if (j < 512) {
      const int f = j >> 5, d = j & 31;
      float q = x[i * 24 + f];
      q = (q - stats[f]) * stats[16 + f] * bng[f] + bnb[f];
      val = q * QnV[f * 32 + d];
    } else {
      const int e = (j - 512) >> 5, d = j & 31;
      const int idx = (int)x[i * 24 + 16 + e] + e * 100;
      val = QlV[(size_t)idx * 32 + d];
    }
    feats[i * IN_DIM + j] = f2bf(val);
  }
}

// ---------------------------------------------------------------------------
// Pack fp32 W (K x 4096, row-major) into bf16 WMMA-B tile layout:
// element p = ((nt*Ktiles + kt)*32 + lane)*16 + j
//          <- W[kt*32 + (lane>>4)*8 + (j&7) + (j>>3)*16][nt*16 + (lane&15)]
// ---------------------------------------------------------------------------
__global__ __launch_bounds__(256)
void pack_w(const float* __restrict__ W, unsigned short* __restrict__ Bp, int Ktiles) {
  const size_t total = (size_t)Ktiles * NT_TILES * 512;
  for (size_t p = (size_t)blockIdx.x * 256 + threadIdx.x; p < total;
       p += (size_t)gridDim.x * 256) {
    const int j    = (int)(p & 15);
    const int lane = (int)((p >> 4) & 31);
    const size_t tile = p >> 9;
    const int kt = (int)(tile % Ktiles);
    const int nt = (int)(tile / Ktiles);
    const int k = kt * 32 + (lane >> 4) * 8 + (j & 7) + (j >> 3) * 16;
    const int n = nt * 16 + (lane & 15);
    Bp[p] = f2bf(W[(size_t)k * GATES + n]);
  }
}

// ---------------------------------------------------------------------------
// A-fragment load per ISA 16-bit A 16x32 layout: per lane two contiguous
// 8-element bf16 chunks (K = base+hf*8 .. +7 and +16..+23) -> two b128 loads.
// ---------------------------------------------------------------------------
__device__ __forceinline__ v16bf load_a_tile(const unsigned short* arow, int hf) {
  const v4u lo = *(const v4u*)(arow + hf * 8);
  const v4u hi = *(const v4u*)(arow + hf * 8 + 16);
  BfPack p;
  p.u[0] = lo.x; p.u[1] = lo.y; p.u[2] = lo.z; p.u[3] = lo.w;
  p.u[4] = hi.x; p.u[5] = hi.y; p.u[6] = hi.z; p.u[7] = hi.w;
  return p.b;
}

// WGP-scope (scope 0, default) prefetch: pulls into all cache levels,
// staging streaming B tiles from L2 into the WGP cache ahead of the loads.
// Used only OUTSIDE the WMMA loop so it cannot perturb the hot schedule.
__device__ __forceinline__ void prefetch_wgp(const void* p) {
  asm volatile("global_prefetch_b8 %0, off" :: "v"(p));
}

// ---------------------------------------------------------------------------
// gates (128 x 4096) = [A0 | A1](128 x K) * Bpacked + bias, bf16 WMMA f32-acc.
// Per wave: 32(M) x 32(N) block = 2 M-tiles x 2 N-tiles, 4 accumulators.
// K loop statically split at the A0/A1 region boundary (both multiples of 32)
// so the hot loop is pure fixed-stride loads + 4 WMMAs.
// ---------------------------------------------------------------------------
__global__ __launch_bounds__(256)
void gates_gemm(const unsigned short* __restrict__ A0, int lda0, int kt0,
                const unsigned short* __restrict__ A1, int lda1, int kt1,
                const unsigned short* __restrict__ Bp,
                const float* __restrict__ bias,
                float* __restrict__ C, int Ktiles) {
  const int tid  = threadIdx.x;
  const int lane = tid & 31;
  const int wave = blockIdx.x * 8 + (tid >> 5);   // 0..511
  const int mb = wave & 3;                        // 4 M blocks of 32 rows
  const int ng = wave >> 2;                       // 128 N groups of 32 cols
  const int hf = lane >> 4;
  const int ml = lane & 15;
  const int m0 = mb * 32 + ml;                    // M-tile 0 row
  const int m1 = m0 + 16;                         // M-tile 1 row

  const v8u* bp0 = (const v8u*)Bp + ((size_t)(ng * 2 + 0) * Ktiles) * 32 + lane;
  const v8u* bp1 = (const v8u*)Bp + ((size_t)(ng * 2 + 1) * Ktiles) * 32 + lane;

  // Stage the head of both B streams into WGP$ (cold-start latency cover);
  // steady state is hidden by 8-wave overlap + pipelined loads.
  #pragma unroll
  for (int pf = 0; pf < 4; ++pf) {
    prefetch_wgp(bp0 + (size_t)pf * 32);
    prefetch_wgp(bp1 + (size_t)pf * 32);
  }

  v8f acc[4] = {};

  auto run = [&](const unsigned short* r0, const unsigned short* r1,
                 const v8u* b0s, const v8u* b1s, int nkt) {
    for (int kt = 0; kt < nkt; ++kt) {
      const v16bf a0 = load_a_tile(r0 + kt * 32, hf);
      const v16bf a1 = load_a_tile(r1 + kt * 32, hf);
      BfPack b0; b0.u = b0s[(size_t)kt * 32];
      BfPack b1; b1.u = b1s[(size_t)kt * 32];
      acc[0] = __builtin_amdgcn_wmma_f32_16x16x32_bf16(false, a0, false, b0.b, (short)0, acc[0], false, false);
      acc[1] = __builtin_amdgcn_wmma_f32_16x16x32_bf16(false, a0, false, b1.b, (short)0, acc[1], false, false);
      acc[2] = __builtin_amdgcn_wmma_f32_16x16x32_bf16(false, a1, false, b0.b, (short)0, acc[2], false, false);
      acc[3] = __builtin_amdgcn_wmma_f32_16x16x32_bf16(false, a1, false, b1.b, (short)0, acc[3], false, false);
    }
  };

  // phase 1: h-region (A0), K tiles [0, kt0)
  run(A0 + (size_t)m0 * lda0, A0 + (size_t)m1 * lda0, bp0, bp1, kt0);
  // phase 2: x-region (A1), K tiles [kt0, kt0+kt1)
  run(A1 + (size_t)m0 * lda1, A1 + (size_t)m1 * lda1,
      bp0 + (size_t)kt0 * 32, bp1 + (size_t)kt0 * 32, kt1);

  // store with bias; C layout: m = half*8 + r, n = lane&15
  #pragma unroll
  for (int i = 0; i < 2; ++i) {
    #pragma unroll
    for (int jn = 0; jn < 2; ++jn) {
      const int col = ng * 32 + jn * 16 + ml;
      const float bv = bias[col];
      const int mbase = mb * 32 + i * 16 + hf * 8;
      const v8f a = acc[i * 2 + jn];
      #pragma unroll
      for (int r = 0; r < 8; ++r)
        C[(size_t)(mbase + r) * GATES + col] = a[r] + bv;
    }
  }
}

// ---------------------------------------------------------------------------
// Per-sample LSTM cell: 4 gate layernorms (block reductions over H=1024),
// activations, state update; writes new h as bf16. Last layer also fuses the
// 1024->2 FC + softmax into d_out.
// ---------------------------------------------------------------------------
__global__ __launch_bounds__(256)
void lstm_update(const float* __restrict__ gates,
                 float* __restrict__ c,
                 unsigned short* __restrict__ hbf,
                 const float* __restrict__ lng,   // this layer's (4,H) gamma
                 const float* __restrict__ lnb,
                 const float* __restrict__ fcW, const float* __restrict__ fcb,
                 float* __restrict__ out, int t, int isLast) {
  const int n = blockIdx.x;
  const int tid = threadIdx.x;
  __shared__ float red[256];
  __shared__ float gmean[4], gscale[4];
  const float* grow = gates + (size_t)n * GATES;

  float sum[4], sq[4];
  #pragma unroll
  for (int g = 0; g < 4; ++g) {
    float s = 0.f, s2 = 0.f;
    #pragma unroll
    for (int jj = 0; jj < 4; ++jj) {
      float v = grow[g * H_DIM + tid + jj * 256];
      s += v; s2 += v * v;
    }
    sum[g] = s; sq[g] = s2;
  }
  #pragma unroll
  for (int g = 0; g < 4; ++g) {
    red[tid] = sum[g]; __syncthreads();
    for (int o = 128; o > 0; o >>= 1) { if (tid < o) red[tid] += red[tid + o]; __syncthreads(); }
    if (tid == 0) gmean[g] = red[0] * (1.f / H_DIM);
    __syncthreads();
    red[tid] = sq[g]; __syncthreads();
    for (int o = 128; o > 0; o >>= 1) { if (tid < o) red[tid] += red[tid + o]; __syncthreads(); }
    if (tid == 0) {
      float m = gmean[g];
      gscale[g] = rsqrtf(red[0] * (1.f / H_DIM) - m * m + 1e-5f);
    }
    __syncthreads();
  }

  float dot0 = 0.f, dot1 = 0.f;
  #pragma unroll
  for (int jj = 0; jj < 4; ++jj) {
    const int j = tid + jj * 256;
    float ig = (grow[j]             - gmean[0]) * gscale[0] * lng[j]             + lnb[j];
    float fg = (grow[H_DIM + j]     - gmean[1]) * gscale[1] * lng[H_DIM + j]     + lnb[H_DIM + j];
    float gn = (grow[2 * H_DIM + j] - gmean[2]) * gscale[2] * lng[2 * H_DIM + j] + lnb[2 * H_DIM + j];
    float og = (grow[3 * H_DIM + j] - gmean[3]) * gscale[3] * lng[3 * H_DIM + j] + lnb[3 * H_DIM + j];
    ig = 1.f / (1.f + __expf(-ig));
    fg = 1.f / (1.f + __expf(-fg));
    og = 1.f / (1.f + __expf(-og));
    float cv = c[(size_t)n * H_DIM + j] * fg + ig * tanhf(gn);
    c[(size_t)n * H_DIM + j] = cv;
    float xv = og * tanhf(cv) + gn;
    hbf[(size_t)n * H_DIM + j] = f2bf(xv);
    if (isLast) { dot0 += xv * fcW[j * 2]; dot1 += xv * fcW[j * 2 + 1]; }
  }
  if (isLast) {
    red[tid] = dot0; __syncthreads();
    for (int o = 128; o > 0; o >>= 1) { if (tid < o) red[tid] += red[tid + o]; __syncthreads(); }
    if (tid == 0) gmean[0] = red[0];
    __syncthreads();
    red[tid] = dot1; __syncthreads();
    for (int o = 128; o > 0; o >>= 1) { if (tid < o) red[tid] += red[tid + o]; __syncthreads(); }
    if (tid == 0) {
      float l0 = gmean[0] + fcb[0];
      float l1 = red[0]   + fcb[1];
      float mx = fmaxf(l0, l1);
      float e0 = __expf(l0 - mx), e1 = __expf(l1 - mx);
      float inv = 1.f / (e0 + e1);
      out[((size_t)n * T_STEPS + t) * 2 + 0] = e0 * inv;
      out[((size_t)n * T_STEPS + t) * 2 + 1] = e1 * inv;
    }
  }
}

__global__ __launch_bounds__(256)
void zero_ws(float* __restrict__ p, size_t nfloat) {
  for (size_t i = (size_t)blockIdx.x * 256 + threadIdx.x; i < nfloat;
       i += (size_t)gridDim.x * 256)
    p[i] = 0.f;
}

extern "C" void kernel_launch(void* const* d_in, const int* in_sizes, int n_in,
                              void* d_out, int out_size, void* d_ws, size_t ws_size,
                              hipStream_t stream) {
  const float* x   = (const float*)d_in[0];
  const float* QnV = (const float*)d_in[1];
  const float* QlV = (const float*)d_in[2];
  const float* bng = (const float*)d_in[3];
  const float* bnb = (const float*)d_in[4];
  const float* W0  = (const float*)d_in[5];
  const float* b0  = (const float*)d_in[6];
  const float* W1  = (const float*)d_in[7];
  const float* b1  = (const float*)d_in[8];
  const float* lng = (const float*)d_in[9];
  const float* lnb = (const float*)d_in[10];
  const float* fcW = (const float*)d_in[11];
  const float* fcb = (const float*)d_in[12];
  float* out = (float*)d_out;

  char* w = (char*)d_ws;
  size_t off = 0;
  float* stats = (float*)(w + off);                   off += 256;
  unsigned short* feats = (unsigned short*)(w + off); off += (size_t)N_BATCH * T_STEPS * IN_DIM * 2;
  unsigned short* pW0   = (unsigned short*)(w + off); off += (size_t)KT_L0 * NT_TILES * 512 * 2;
  unsigned short* pW1   = (unsigned short*)(w + off); off += (size_t)KT_L1 * NT_TILES * 512 * 2;
  float* gbuf = (float*)(w + off);                    off += (size_t)N_BATCH * GATES * 4;
  float* c0   = (float*)(w + off);                    off += (size_t)N_BATCH * H_DIM * 4;
  float* c1   = (float*)(w + off);                    off += (size_t)N_BATCH * H_DIM * 4;
  unsigned short* h0 = (unsigned short*)(w + off);    off += (size_t)N_BATCH * H_DIM * 2;
  unsigned short* h1 = (unsigned short*)(w + off);    off += (size_t)N_BATCH * H_DIM * 2;

  // c0,c1,h0,h1 are contiguous: 2*128K floats + 256K bf16 = 393216 floats
  zero_ws<<<1536, 256, 0, stream>>>(c0, (size_t)393216);

  bn_stats<<<16, 256, 0, stream>>>(x, stats);
  build_feats<<<N_BATCH * T_STEPS, 256, 0, stream>>>(x, QnV, QlV, bng, bnb, stats, feats);
  pack_w<<<4096, 256, 0, stream>>>(W0, pW0, KT_L0);
  pack_w<<<4096, 256, 0, stream>>>(W1, pW1, KT_L1);

  for (int t = 0; t < T_STEPS; ++t) {
    // layer 0: A = [h0_prev (32 K-tiles) | feats[:,t,:] (24 K-tiles)]
    gates_gemm<<<64, 256, 0, stream>>>(h0, H_DIM, 32,
                                       feats + (size_t)t * IN_DIM, T_STEPS * IN_DIM, 24,
                                       pW0, b0, gbuf, KT_L0);
    lstm_update<<<128, 256, 0, stream>>>(gbuf, c0, h0, lng, lnb, fcW, fcb, out, t, 0);
    // layer 1: A = [h1_prev (32 K-tiles) | h0_new (32 K-tiles)]
    gates_gemm<<<64, 256, 0, stream>>>(h1, H_DIM, 32,
                                       h0, H_DIM, 32,
                                       pW1, b1, gbuf, KT_L1);
    lstm_update<<<128, 256, 0, stream>>>(gbuf, c1, h1, lng + 4 * H_DIM, lnb + 4 * H_DIM,
                                         fcW, fcb, out, t, 1);
  }
}
